// Attention_42537356099868
// MI455X (gfx1250) — compile-verified, matrix operand order checked by reading, not run on hardware
//
#include <hip/hip_runtime.h>
#include <hip/hip_bf16.h>
#include <cstdint>

// ---------------- problem constants ----------------
#define B_DIM 4
#define T_DIM 2048
#define C_DIM 768
#define H_DIM 12
#define HD_DIM 64
#define N3 (3 * C_DIM)          // 2304
#define MROWS (B_DIM * T_DIM)   // 8192

// ---------------- CDNA5 WMMA types ----------------
typedef __attribute__((ext_vector_type(16))) __bf16        v16bf;
typedef __attribute__((ext_vector_type(8)))  float         v8f;
typedef __attribute__((ext_vector_type(4)))  unsigned int  u32x4;

union Frag {
  v16bf v;
  u32x4 u[2];
  unsigned short s[16];
};

__device__ inline unsigned short f32_to_bf16(float f) {
  unsigned int u = __float_as_uint(f);
  unsigned int r = 0x7FFFu + ((u >> 16) & 1u);
  return (unsigned short)((u + r) >> 16);
}

// Load a 16x32 bf16 A-fragment (or B-fragment from a [N,K]-major buffer).
// Lane layout (ISA 7.12.2): lane holds row (lane&15); element e holds
// K = (e/8)*16 + (lane>>4)*8 + (e%8)  -> two contiguous 16-byte loads.
__device__ inline v16bf load_frag(const unsigned short* p, int row, int ld,
                                  int k0, int lane) {
  const int half8 = (lane >> 4) << 3;           // 0 or 8
  const unsigned short* q = p + (size_t)row * ld + k0 + half8;
  Frag f;
  f.u[0] = *(const u32x4*)(q);                  // K = k0 + half8 .. +7
  f.u[1] = *(const u32x4*)(q + 16);             // K = k0 + 16 + half8 .. +7
  return f.v;
}

__device__ inline v8f wmma_bf16(v16bf a, v16bf b, v8f c) {
  return __builtin_amdgcn_wmma_f32_16x16x32_bf16(
      /*neg_a=*/false, a, /*neg_b=*/false, b,
      /*c_mod=*/(short)0, c, /*reuse_a=*/false, /*reuse_b=*/false);
}

// CDNA5 async global->LDS copy (ASYNCcnt path). lds = byte offset into LDS
// (low 32 bits of a generic pointer to __shared__), g = global source.
__device__ inline void async_copy_b128(unsigned int lds, const unsigned short* g) {
  asm volatile("global_load_async_to_lds_b128 %0, %1, off"
               :: "v"(lds), "v"(g) : "memory");
}

// xor-style lane swap within each 16-lane half via v_permlane16_b32 (VALU,
// no LDS round-trip). Selector packs 16 4-bit source-lane indices.
__device__ inline float permlane16_xor(float v, unsigned sel_lo, unsigned sel_hi) {
  unsigned u = __float_as_uint(v);
  unsigned r = __builtin_amdgcn_permlane16(u, u, sel_lo, sel_hi, false, false);
  return __uint_as_float(r);
}

__device__ inline float rowmax16(float v) {
  v = fmaxf(v, permlane16_xor(v, 0x67452301u, 0xEFCDAB89u));  // xor 1
  v = fmaxf(v, permlane16_xor(v, 0x54761032u, 0xDCFE98BAu));  // xor 2
  v = fmaxf(v, permlane16_xor(v, 0x32107654u, 0xBA98FEDCu));  // xor 4
  v = fmaxf(v, permlane16_xor(v, 0xFEDCBA98u, 0x76543210u));  // xor 8
  return v;
}

// ---------------- conversion / layout kernels ----------------
__global__ void cvt_f32_bf16_kernel(const float* __restrict__ in,
                                    unsigned short* __restrict__ out, int n) {
  int i = blockIdx.x * blockDim.x + threadIdx.x;
  if (i < n) out[i] = f32_to_bf16(in[i]);
}

// wT[n*K + k] = bf16( w[k*N + n] )   (w is [K,N] row-major)
__global__ void transpose_w_kernel(const float* __restrict__ w,
                                   unsigned short* __restrict__ wT,
                                   int K, int N) {
  int i = blockIdx.x * blockDim.x + threadIdx.x;
  if (i < K * N) {
    int n = i / K, k = i % K;
    wT[i] = f32_to_bf16(w[(size_t)k * N + n]);
  }
}

// vt[((b*H + h)*HD + d)*T + t] = qkv[(b*T + t)*N3 + 2C + h*HD + d]
__global__ void transpose_v_kernel(const unsigned short* __restrict__ qkvb,
                                   unsigned short* __restrict__ vt) {
  int i = blockIdx.x * blockDim.x + threadIdx.x;
  if (i >= B_DIM * H_DIM * HD_DIM * T_DIM) return;
  int t = i % T_DIM;
  int rest = i / T_DIM;
  int d = rest % HD_DIM;
  int bh = rest / HD_DIM;
  int h = bh % H_DIM, b = bh / H_DIM;
  vt[i] = qkvb[(size_t)(b * T_DIM + t) * N3 + 2 * C_DIM + h * HD_DIM + d];
}

// ---------------- fused GEMM (A[M,K]bf16 @ Bt[N,K]bf16 + bias) -------------
// Block = 4 waves, block tile 128(M) x 64(N). The 64x32 B k-chunk is shared
// by all waves: staged into LDS with double-buffered async global->LDS
// copies (2x b128 per lane per chunk). Each wave computes 32x64 via
// 8 WMMAs per K-step (2 A-frags x 4 LDS B-frags).
template <int NOUT, bool BF16OUT>
__global__ __launch_bounds__(128) void gemm_kernel(
    const unsigned short* __restrict__ A,    // [MROWS, C] bf16
    const unsigned short* __restrict__ Bt,   // [NOUT, C] bf16 (transposed)
    const float* __restrict__ bias,          // [NOUT]
    void* __restrict__ outp) {               // [MROWS, NOUT] bf16 or f32
  constexpr int KD = C_DIM;
  constexpr int KSTEPS = KD / 32;            // 24
  __shared__ __align__(16) unsigned short btile[2][64][32];  // 2 x 4KB

  const int lane = threadIdx.x & 31;
  const int wave = threadIdx.x >> 5;
  const int nl = lane & 15, half = lane >> 4;

  const int col0 = blockIdx.x * 64;
  const int row0 = blockIdx.y * 128 + wave * 32;

  // staging map: thread -> (B column c, 32-byte k-segment seg)
  const int c = threadIdx.x >> 1;            // 0..63
  const int seg = threadIdx.x & 1;           // 0..1
  const unsigned short* gB = Bt + (size_t)(col0 + c) * KD + seg * 16;
  const unsigned int lds0 = (unsigned int)(size_t)&btile[0][c][seg * 16];
  const unsigned int lds1 = (unsigned int)(size_t)&btile[1][c][seg * 16];

  v8f acc[2][4];
#pragma unroll
  for (int mi = 0; mi < 2; ++mi)
#pragma unroll
    for (int j = 0; j < 4; ++j)
#pragma unroll
      for (int r = 0; r < 8; ++r) acc[mi][j][r] = 0.0f;

  // prologue: stage k-chunk 0 into buffer 0
  async_copy_b128(lds0, gB);
  async_copy_b128(lds0 + 16, gB + 8);

#pragma unroll 2
  for (int kc = 0; kc < KSTEPS; ++kc) {
    const int cur = kc & 1;
    const unsigned int ldsn = cur ? lds0 : lds1;
    if (kc + 1 < KSTEPS) {
      const unsigned short* gn = gB + (kc + 1) * 32;
      async_copy_b128(ldsn, gn);
      async_copy_b128(ldsn + 16, gn + 8);
      asm volatile("s_wait_asynccnt 0x2" ::: "memory");  // current chunk landed
    } else {
      asm volatile("s_wait_asynccnt 0x0" ::: "memory");
    }
    __syncthreads();                          // staging visible to all waves

    v16bf a0 = load_frag(A, row0 + nl, KD, kc * 32, lane);
    v16bf a1 = load_frag(A, row0 + 16 + nl, KD, kc * 32, lane);
#pragma unroll
    for (int j = 0; j < 4; ++j) {
      v16bf b = load_frag(&btile[cur][0][0], j * 16 + nl, 32, 0, lane);
      acc[0][j] = wmma_bf16(a0, b, acc[0][j]);
      acc[1][j] = wmma_bf16(a1, b, acc[1][j]);
    }
    __syncthreads();                          // reads done before next overwrite
  }

  // C/D layout: lane holds column n; VGPR r holds row r + 8*(lane/16).
#pragma unroll
  for (int mi = 0; mi < 2; ++mi)
#pragma unroll
    for (int j = 0; j < 4; ++j) {
      int n = col0 + j * 16 + nl;
      float bv = bias[n];
#pragma unroll
      for (int r = 0; r < 8; ++r) {
        int m = row0 + mi * 16 + r + half * 8;
        float v = acc[mi][j][r] + bv;
        if constexpr (BF16OUT)
          ((unsigned short*)outp)[(size_t)m * NOUT + n] = f32_to_bf16(v);
        else
          ((float*)outp)[(size_t)m * NOUT + n] = v;
      }
    }
}

// ---------------- flash attention ----------------
// One wave per (b, h, 16-query tile). Online softmax over key blocks of 32.
// Row-max via v_permlane16 (VALU); row-sum via an extra WMMA against a
// constant ones B-matrix (layout-compatible with the per-lane row stats).
__global__ __launch_bounds__(128) void flash_attn_kernel(
    const unsigned short* __restrict__ qkvb,    // [MROWS, N3] bf16
    const unsigned short* __restrict__ vt,      // [B,H,HD,T]  bf16
    unsigned short* __restrict__ attnb) {       // [MROWS, C]  bf16
  __shared__ __align__(16) unsigned short pbuf[4][16 * 32];  // P transpose, 1KB/wave

  const int lane = threadIdx.x & 31;
  const int wave = threadIdx.x >> 5;
  const int nl = lane & 15, half = lane >> 4;

  int idx = blockIdx.x * 4 + wave;              // 4*12*128 = 6144 waves
  int qt = idx % (T_DIM / 16);
  int bh = idx / (T_DIM / 16);
  int h = bh % H_DIM, b = bh / H_DIM;
  int qbase = qt * 16;

  const unsigned short* base = qkvb + (size_t)b * T_DIM * N3;
  const unsigned short* vbase = vt + (size_t)(b * H_DIM + h) * HD_DIM * T_DIM;

  // Q fragments: 16 rows x hd=64, two K=32 chunks. Column base h*64 (+C for K).
  v16bf qf0 = load_frag(base, qbase + nl, N3, h * HD_DIM + 0, lane);
  v16bf qf1 = load_frag(base, qbase + nl, N3, h * HD_DIM + 32, lane);

  // constant ones B-fragment (bf16 1.0 = 0x3F80) for WMMA row-sum
  Frag ones;
#pragma unroll
  for (int i = 0; i < 16; ++i) ones.s[i] = 0x3F80;

  v8f o[4];
  float mrow[8], lrow[8];
#pragma unroll
  for (int f = 0; f < 4; ++f)
#pragma unroll
    for (int r = 0; r < 8; ++r) o[f][r] = 0.0f;
#pragma unroll
  for (int r = 0; r < 8; ++r) { mrow[r] = -__builtin_inff(); lrow[r] = 0.0f; }

  unsigned short* pb = &pbuf[wave][0];
  const float scale = 0.125f;                   // 1/sqrt(64)
  int nblocks = (qbase + 16 + 31) / 32;

  for (int blk = 0; blk < nblocks; ++blk) {
    int j0 = blk * 32;
    v8f s[2];
    // ---- S = (Q K^T) * scale, with causal mask ----
#pragma unroll
    for (int st = 0; st < 2; ++st) {
#pragma unroll
      for (int r = 0; r < 8; ++r) s[st][r] = 0.0f;
      int keycol = j0 + st * 16 + nl;           // B-frag column = key index
      v16bf kf0 = load_frag(base, keycol, N3, C_DIM + h * HD_DIM + 0, lane);
      v16bf kf1 = load_frag(base, keycol, N3, C_DIM + h * HD_DIM + 32, lane);
      s[st] = wmma_bf16(qf0, kf0, s[st]);
      s[st] = wmma_bf16(qf1, kf1, s[st]);
#pragma unroll
      for (int r = 0; r < 8; ++r) {
        int q = qbase + r + half * 8;
        float v = s[st][r] * scale;
        s[st][r] = (keycol <= q) ? v : -__builtin_inff();
      }
    }
    // ---- online softmax row max (rows live across 16-lane halves) ----
    float mnew[8], alpha[8];
#pragma unroll
    for (int r = 0; r < 8; ++r) {
      float tmax = rowmax16(fmaxf(s[0][r], s[1][r]));
      mnew[r] = fmaxf(mrow[r], tmax);
      alpha[r] = __expf(mrow[r] - mnew[r]);
      mrow[r] = mnew[r];
    }
    // ---- P = exp(S - m); stash bf16 P into LDS for layout transpose ----
#pragma unroll
    for (int st = 0; st < 2; ++st)
#pragma unroll
      for (int r = 0; r < 8; ++r) {
        float p = __expf(s[st][r] - mnew[r]);
        pb[(r + half * 8) * 32 + st * 16 + nl] = f32_to_bf16(p);
      }
    // rescale accumulators
#pragma unroll
    for (int f = 0; f < 4; ++f)
#pragma unroll
      for (int r = 0; r < 8; ++r) o[f][r] *= alpha[r];

    // same-wave LDS RAW: wait for the ds stores before re-reading as A-layout
    asm volatile("s_wait_dscnt 0x0" ::: "memory");
    Frag pf;
    {
      const unsigned short* pr = pb + nl * 32 + (half << 3);
      pf.u[0] = *(const u32x4*)(pr);            // keys (half*8)..+7
      pf.u[1] = *(const u32x4*)(pr + 16);       // keys 16+(half*8)..+7
    }
    // ---- row-sum(P) via WMMA against ones: D[m][n] = sum_k P[m][k] ----
    {
      v8f rs;
#pragma unroll
      for (int r = 0; r < 8; ++r) rs[r] = 0.0f;
      rs = wmma_bf16(pf.v, ones.v, rs);
#pragma unroll
      for (int r = 0; r < 8; ++r) lrow[r] = lrow[r] * alpha[r] + rs[r];
    }
    // ---- O += P @ V : V^T buffer gives contiguous-K B-fragments ----
#pragma unroll
    for (int f = 0; f < 4; ++f) {
      v16bf vf = load_frag(vbase, f * 16 + nl, T_DIM, j0, lane);
      o[f] = wmma_bf16(pf.v, vf, o[f]);
    }
  }
  // ---- normalize and store [B,T,C] bf16 ----
#pragma unroll
  for (int r = 0; r < 8; ++r) lrow[r] = 1.0f / lrow[r];
#pragma unroll
  for (int f = 0; f < 4; ++f) {
    int d = h * HD_DIM + f * 16 + nl;
#pragma unroll
    for (int r = 0; r < 8; ++r) {
      int m = qbase + r + half * 8;
      attnb[(size_t)(b * T_DIM + m) * C_DIM + d] = f32_to_bf16(o[f][r] * lrow[r]);
    }
  }
}

// ---------------- launch ----------------
extern "C" void kernel_launch(void* const* d_in, const int* in_sizes, int n_in,
                              void* d_out, int out_size, void* d_ws, size_t ws_size,
                              hipStream_t stream) {
  const float* x      = (const float*)d_in[0];
  const float* w_attn = (const float*)d_in[1];
  const float* b_attn = (const float*)d_in[2];
  const float* w_proj = (const float*)d_in[3];
  const float* b_proj = (const float*)d_in[4];
  float* out = (float*)d_out;

  // workspace layout (bf16 stored as ushort), ~80 MB total
  unsigned short* ws      = (unsigned short*)d_ws;
  unsigned short* xb      = ws;                                       // 8192*768
  unsigned short* wattnT  = xb     + (size_t)MROWS * C_DIM;           // 2304*768
  unsigned short* wprojT  = wattnT + (size_t)N3 * C_DIM;              // 768*768
  unsigned short* qkvb    = wprojT + (size_t)C_DIM * C_DIM;           // 8192*2304
  unsigned short* vtb     = qkvb   + (size_t)MROWS * N3;              // 4*12*64*2048
  unsigned short* attnb   = vtb    + (size_t)B_DIM * H_DIM * HD_DIM * T_DIM; // 8192*768

  const int nX = MROWS * C_DIM;                 // 6291456
  cvt_f32_bf16_kernel<<<(nX + 255) / 256, 256, 0, stream>>>(x, xb, nX);
  transpose_w_kernel<<<(N3 * C_DIM + 255) / 256, 256, 0, stream>>>(w_attn, wattnT, C_DIM, N3);
  transpose_w_kernel<<<(C_DIM * C_DIM + 255) / 256, 256, 0, stream>>>(w_proj, wprojT, C_DIM, C_DIM);

  // QKV GEMM: block tile 128x64, grid (N3/64, MROWS/128)
  gemm_kernel<N3, true><<<dim3(N3 / 64, MROWS / 128), 128, 0, stream>>>(
      xb, wattnT, b_attn, qkvb);

  transpose_v_kernel<<<(B_DIM * H_DIM * HD_DIM * T_DIM + 255) / 256, 256, 0, stream>>>(qkvb, vtb);

  // flash attention: 4*12*128 waves, 4 waves/block
  flash_attn_kernel<<<B_DIM * H_DIM * (T_DIM / 16) / 4, 128, 0, stream>>>(qkvb, vtb, attnb);

  // projection GEMM: block tile 128x64, grid (C/64, MROWS/128)
  gemm_kernel<C_DIM, false><<<dim3(C_DIM / 64, MROWS / 128), 128, 0, stream>>>(
      attnb, wprojT, b_proj, out);
}
